// Attention_2027224563985
// MI455X (gfx1250) — matrix operand order repaired and checked by reading, not compiled, not run on hardware
//
#include <hip/hip_runtime.h>

typedef unsigned short u16;
typedef __bf16 v16bf __attribute__((ext_vector_type(16)));
typedef float  v8f   __attribute__((ext_vector_type(8)));

#define BATCH 4
#define SEQ   2048
#define DIM   1024
#define MINF  (-3.402823466e38f)
#define SCALE 0.02209708691207961f   /* 1/sqrt(2048): reference scales by sqrt(kT.shape[-1]) = sqrt(N) */

// ---------- helpers ----------
__device__ __forceinline__ u16 f2bf(float f) {
    union { float f; unsigned u; } x; x.f = f;
    unsigned u = x.u;
    unsigned r = (u + 0x7fffu + ((u >> 16) & 1u)) >> 16;   // round-to-nearest-even
    return (u16)r;
}

// A-fragment (16x32 bf16, M x K): lane m = lane&15, kHalf = (lane>>4)*8.
// two 16B loads: K=[kHalf..kHalf+8) and K=[16+kHalf..16+kHalf+8)
__device__ __forceinline__ v16bf load_frag_pair(const u16* p0, const u16* p1) {
    union { v16bf v; uint4 q[2]; } u;
    u.q[0] = *(const uint4*)p0;
    u.q[1] = *(const uint4*)p1;
    return u.v;
}
// B-fragment (32x16 bf16, K x N): lane n = lane&15, K=[(lane>>4)*16 .. +16) contiguous (32B)
__device__ __forceinline__ v16bf load_frag_b(const u16* p) {
    union { v16bf v; uint4 q[2]; } u;
    u.q[0] = *(const uint4*)p;
    u.q[1] = *(const uint4*)(p + 8);
    return u.v;
}

// ---------- conversion kernels ----------
__global__ void cvt_bf16_kernel(const float* __restrict__ in, u16* __restrict__ out, int n) {
    int i = blockIdx.x * blockDim.x + threadIdx.x;
    if (i < n) out[i] = f2bf(in[i]);
}

// WT[n][k] = W[k][n], both DIM x DIM
__global__ void transpose_bf16_kernel(const float* __restrict__ W, u16* __restrict__ WT) {
    int i = blockIdx.x * blockDim.x + threadIdx.x;   // i = k*DIM + n
    int n = i & (DIM - 1);
    int k = i >> 10;
    WT[(size_t)n * DIM + k] = f2bf(W[i]);
}

// ---------- bf16 GEMM, 32x64 per wave: C[M x Nc] = A[M x K] * Bt[Nc x K]^T + bias ----------
// grid: (Nc/256, M/32), block 128 (4 waves). Wave computes 2x4 grid of 16x16 frags.
// mode 0: bf16 row-major out (Q,K) | mode 1: bf16 Vt[b][d][tok] | mode 2: f32 row-major (-> d_out)
__global__ __launch_bounds__(128) void gemm_bf16_kernel(
        const u16* __restrict__ A, const u16* __restrict__ Bt,
        const float* __restrict__ bias, void* __restrict__ out,
        int M, int Nc, int K, int mode) {
    int lane  = threadIdx.x & 31;
    int wave  = threadIdx.x >> 5;
    int ln = lane & 15, hi = lane >> 4;

    int mbase = blockIdx.y * 32;
    int nbase = (blockIdx.x * 4 + wave) * 64;

    const u16* arow0 = A + (size_t)(mbase + ln) * K;
    const u16* arow1 = A + (size_t)(mbase + 16 + ln) * K;
    const u16* brow0 = Bt + (size_t)(nbase + 0 * 16 + ln) * K;
    const u16* brow1 = Bt + (size_t)(nbase + 1 * 16 + ln) * K;
    const u16* brow2 = Bt + (size_t)(nbase + 2 * 16 + ln) * K;
    const u16* brow3 = Bt + (size_t)(nbase + 3 * 16 + ln) * K;

    v8f c[2][4];
#pragma unroll
    for (int mi = 0; mi < 2; ++mi)
#pragma unroll
        for (int nj = 0; nj < 4; ++nj)
            c[mi][nj] = (v8f){0.f,0.f,0.f,0.f,0.f,0.f,0.f,0.f};

    for (int kk = 0; kk < K; kk += 32) {
        // stream prefetch 256B ahead of each operand stream (-> global_prefetch_b8)
        __builtin_prefetch(arow0 + kk + 128, 0, 1);
        __builtin_prefetch(arow1 + kk + 128, 0, 1);
        __builtin_prefetch(brow0 + kk + 128, 0, 1);
        __builtin_prefetch(brow2 + kk + 128, 0, 1);

        v16bf a0 = load_frag_pair(arow0 + kk + hi * 16, arow0 + kk + hi * 16 + 8);
        v16bf a1 = load_frag_pair(arow1 + kk + hi * 16, arow1 + kk + hi * 16 + 8);
        v16bf b0 = load_frag_b(brow0 + kk + hi * 16);
        v16bf b1 = load_frag_b(brow1 + kk + hi * 16);
        v16bf b2 = load_frag_b(brow2 + kk + hi * 16);
        v16bf b3 = load_frag_b(brow3 + kk + hi * 16);
        c[0][0] = __builtin_amdgcn_wmma_f32_16x16x32_bf16(false, a0, false, b0, (short)0, c[0][0], false, false);
        c[1][0] = __builtin_amdgcn_wmma_f32_16x16x32_bf16(false, a1, false, b0, (short)0, c[1][0], false, false);
        c[0][1] = __builtin_amdgcn_wmma_f32_16x16x32_bf16(false, a0, false, b1, (short)0, c[0][1], false, false);
        c[1][1] = __builtin_amdgcn_wmma_f32_16x16x32_bf16(false, a1, false, b1, (short)0, c[1][1], false, false);
        c[0][2] = __builtin_amdgcn_wmma_f32_16x16x32_bf16(false, a0, false, b2, (short)0, c[0][2], false, false);
        c[1][2] = __builtin_amdgcn_wmma_f32_16x16x32_bf16(false, a1, false, b2, (short)0, c[1][2], false, false);
        c[0][3] = __builtin_amdgcn_wmma_f32_16x16x32_bf16(false, a0, false, b3, (short)0, c[0][3], false, false);
        c[1][3] = __builtin_amdgcn_wmma_f32_16x16x32_bf16(false, a1, false, b3, (short)0, c[1][3], false, false);
    }

    // epilogue: uniform mode, branch hoisted outside the store loops
    if (mode == 0) {
        u16* o = (u16*)out;
#pragma unroll
        for (int nj = 0; nj < 4; ++nj) {
            int col = nbase + nj * 16 + ln;
            float bv = bias[col];
#pragma unroll
            for (int mi = 0; mi < 2; ++mi)
#pragma unroll
                for (int r = 0; r < 8; ++r) {
                    int row = mbase + mi * 16 + r + hi * 8;
                    o[(size_t)row * Nc + col] = f2bf(c[mi][nj][r] + bv);
                }
        }
    } else if (mode == 1) {                       // Vt[b][d][tok]
        u16* o = (u16*)out;
#pragma unroll
        for (int nj = 0; nj < 4; ++nj) {
            int col = nbase + nj * 16 + ln;
            float bv = bias[col];
#pragma unroll
            for (int mi = 0; mi < 2; ++mi)
#pragma unroll
                for (int r = 0; r < 8; ++r) {
                    int row = mbase + mi * 16 + r + hi * 8;
                    int bb = row >> 11;           // / SEQ
                    int tn = row & (SEQ - 1);
                    o[((size_t)bb * DIM + col) * SEQ + tn] = f2bf(c[mi][nj][r] + bv);
                }
        }
    } else {
        float* o = (float*)out;
#pragma unroll
        for (int nj = 0; nj < 4; ++nj) {
            int col = nbase + nj * 16 + ln;
            float bv = bias[col];
#pragma unroll
            for (int mi = 0; mi < 2; ++mi)
#pragma unroll
                for (int r = 0; r < 8; ++r) {
                    int row = mbase + mi * 16 + r + hi * 8;
                    o[(size_t)row * Nc + col] = c[mi][nj][r] + bv;
                }
        }
    }
}

// ---------- flash attention: ctx[tok][d] (bf16) ----------
// grid.x = BATCH*SEQ/16 ; block = 256 (8 waves).
// Every wave: partial S over its 128-wide d-slice (Q frags register-resident) ->
// LDS f32 reduce (ds_add_f32), redundant per-wave softmax straight from LDS
// (yields P in A-fragment layout), then P*V WMMAs for its 128-wide d-slice of O.
__global__ __launch_bounds__(256) void attn_kernel(
        const u16* __restrict__ Qb, const u16* __restrict__ Kb, const u16* __restrict__ Vt,
        const unsigned char* __restrict__ mask, u16* __restrict__ ctx) {
    __shared__ float ldsS[16 * 32];               // full-precision score tile (16 q-rows x 32 keys)

    int blk  = blockIdx.x;
    int b    = blk >> 7;                          // / (SEQ/16)
    int qblk = blk & 127;
    int tid  = threadIdx.x;
    int lane = tid & 31;
    int wave = tid >> 5;
    int ln = lane & 15, hi = lane >> 4;

    size_t qrow0 = (size_t)b * SEQ + qblk * 16;   // global token row of this query block
    int dsl = wave * 128;                         // this wave's d-slice

    v8f O[8];
#pragma unroll
    for (int j = 0; j < 8; ++j) O[j] = (v8f){0.f,0.f,0.f,0.f,0.f,0.f,0.f,0.f};

    float mrow = MINF, lrow = 0.f;                // running stats for row (lane&15)

    // Q fragments for this wave's d-slice: resident in registers for the whole key loop
    const u16* arow = Qb + (qrow0 + ln) * DIM;
    v16bf qf[4];
#pragma unroll
    for (int t = 0; t < 4; ++t)
        qf[t] = load_frag_pair(arow + dsl + t * 32 + hi * 8,
                               arow + dsl + t * 32 + 16 + hi * 8);

    const unsigned char* mrowp = mask + ((size_t)b * SEQ + qblk * 16 + ln) * SEQ;
    size_t krowbase = (size_t)b * SEQ;
    const u16* vbase = Vt + (size_t)b * DIM * SEQ;

    for (int kb = 0; kb < SEQ / 32; ++kb) {
        // prefetch next key block's K/V streams for this d-slice
        if (kb + 1 < SEQ / 32) {
            __builtin_prefetch(Kb + (krowbase + (kb + 1) * 32 + ln) * DIM + dsl, 0, 1);
            __builtin_prefetch(Kb + (krowbase + (kb + 1) * 32 + 16 + ln) * DIM + dsl, 0, 1);
            __builtin_prefetch(vbase + ((size_t)(dsl + ln)) * SEQ + (kb + 1) * 32, 0, 1);
            __builtin_prefetch(vbase + ((size_t)(dsl + 64 + ln)) * SEQ + (kb + 1) * 32, 0, 1);
        }

        // --- partial S over d-slice [dsl, dsl+128) ---
        v8f S0 = (v8f){0.f,0.f,0.f,0.f,0.f,0.f,0.f,0.f};
        v8f S1 = S0;
        {
            const u16* b0 = Kb + (krowbase + kb * 32 + ln) * DIM;
            const u16* b1 = Kb + (krowbase + kb * 32 + 16 + ln) * DIM;
#pragma unroll
            for (int t = 0; t < 4; ++t) {
                int dk = dsl + t * 32;
                v16bf f0 = load_frag_pair(b0 + dk + hi * 8, b0 + dk + 16 + hi * 8);
                v16bf f1 = load_frag_pair(b1 + dk + hi * 8, b1 + dk + 16 + hi * 8);
                S0 = __builtin_amdgcn_wmma_f32_16x16x32_bf16(false, qf[t], false, f0, (short)0, S0, false, false);
                S1 = __builtin_amdgcn_wmma_f32_16x16x32_bf16(false, qf[t], false, f1, (short)0, S1, false, false);
            }
        }
        __syncthreads();                          // previous iteration's LDS reads complete
        ldsS[tid]       = 0.f;
        ldsS[tid + 256] = 0.f;
        __syncthreads();
#pragma unroll
        for (int r = 0; r < 8; ++r) {             // C-layout: row r+hi*8, cols ln / 16+ln
            atomicAdd(&ldsS[(r + hi * 8) * 32 + ln],      S0[r]);
            atomicAdd(&ldsS[(r + hi * 8) * 32 + 16 + ln], S1[r]);
        }
        __syncthreads();

        // --- redundant per-wave softmax; lane owns row ln, K chunk hi*8 (A-frag layout) ---
        unsigned long long mbits0 = *(const unsigned long long*)(mrowp + kb * 32 + hi * 8);
        unsigned long long mbits1 = *(const unsigned long long*)(mrowp + kb * 32 + 16 + hi * 8);
        float s0[8], s1[8];
        float smax = MINF;
#pragma unroll
        for (int j = 0; j < 8; ++j) {
            float v0 = ldsS[ln * 32 + hi * 8 + j] * SCALE;
            float v1 = ldsS[ln * 32 + 16 + hi * 8 + j] * SCALE;
            s0[j] = ((mbits0 >> (8 * j)) & 1ull) ? MINF : v0;
            s1[j] = ((mbits1 >> (8 * j)) & 1ull) ? MINF : v1;
            smax = fmaxf(smax, fmaxf(s0[j], s1[j]));
        }
        smax = fmaxf(smax, __shfl_xor(smax, 16, 32));     // row's 32 keys live in lanes {ln, ln+16}
        float mn    = fmaxf(mrow, smax);
        float alpha = __expf(mrow - mn);
        union { v16bf v; u16 h[16]; } P;
        float rs = 0.f;
#pragma unroll
        for (int j = 0; j < 8; ++j) {
            float p0 = __expf(s0[j] - mn);
            float p1 = __expf(s1[j] - mn);
            P.h[j]     = f2bf(p0);
            P.h[8 + j] = f2bf(p1);
            rs += p0 + p1;
        }
        rs += __shfl_xor(rs, 16, 32);
        lrow = lrow * alpha + rs;
        mrow = mn;

        // --- O_slice = O_slice * alpha_row + P * V_block ---
#pragma unroll
        for (int j = 0; j < 8; ++j) {
            int dcol = dsl + j * 16 + ln;
            const u16* vp = vbase + (size_t)dcol * SEQ + kb * 32 + hi * 8;
            v16bf vf = load_frag_pair(vp, vp + 16);
#pragma unroll
            for (int r = 0; r < 8; ++r) {
                float a_r = __shfl(alpha, r + hi * 8, 32);  // alpha for C-layout row r+hi*8
                O[j][r] *= a_r;
            }
            O[j] = __builtin_amdgcn_wmma_f32_16x16x32_bf16(false, P.v, false, vf, (short)0, O[j], false, false);
        }
    }

    float linv = 1.0f / lrow;                     // for row ln
#pragma unroll
    for (int j = 0; j < 8; ++j) {
        int dcol = dsl + j * 16 + ln;
#pragma unroll
        for (int r = 0; r < 8; ++r) {
            float li = __shfl(linv, r + hi * 8, 32);
            size_t row = qrow0 + r + hi * 8;
            ctx[row * DIM + dcol] = f2bf(O[j][r] * li);
        }
    }
}

// ---------- workspace layout (bytes) ----------
#define OFF_XB   ((size_t)0)                      // 16 MB  bf16 x [B*SEQ][DIM]
#define OFF_WQT  ((size_t)16777216)               //  2 MB
#define OFF_WKT  ((size_t)18874368)
#define OFF_WVT  ((size_t)20971520)
#define OFF_WOT  ((size_t)23068672)
#define OFF_QB   ((size_t)25165824)               // 16 MB
#define OFF_KB   ((size_t)41943040)               // 16 MB
#define OFF_VT   ((size_t)58720256)               // 16 MB  Vt[b][d][tok]
#define OFF_CTX  ((size_t)75497472)               // 16 MB
// total 92274688 bytes (~88 MB)

extern "C" void kernel_launch(void* const* d_in, const int* in_sizes, int n_in,
                              void* d_out, int out_size, void* d_ws, size_t ws_size,
                              hipStream_t stream) {
    const float* x  = (const float*)d_in[0];
    const unsigned char* mask = (const unsigned char*)d_in[1];
    const float* Wq = (const float*)d_in[2];
    const float* bq = (const float*)d_in[3];
    const float* Wk = (const float*)d_in[4];
    const float* bk = (const float*)d_in[5];
    const float* Wv = (const float*)d_in[6];
    const float* bv = (const float*)d_in[7];
    const float* Wo = (const float*)d_in[8];
    const float* bo = (const float*)d_in[9];

    char* ws = (char*)d_ws;
    u16* xb  = (u16*)(ws + OFF_XB);
    u16* WqT = (u16*)(ws + OFF_WQT);
    u16* WkT = (u16*)(ws + OFF_WKT);
    u16* WvT = (u16*)(ws + OFF_WVT);
    u16* WoT = (u16*)(ws + OFF_WOT);
    u16* Qb  = (u16*)(ws + OFF_QB);
    u16* Kb  = (u16*)(ws + OFF_KB);
    u16* Vt  = (u16*)(ws + OFF_VT);
    u16* Ctx = (u16*)(ws + OFF_CTX);

    const int Mtok = BATCH * SEQ;                 // 8192
    const int nx = Mtok * DIM;                    // 8388608

    cvt_bf16_kernel<<<(nx + 255) / 256, 256, 0, stream>>>(x, xb, nx);
    transpose_bf16_kernel<<<(DIM * DIM) / 256, 256, 0, stream>>>(Wq, WqT);
    transpose_bf16_kernel<<<(DIM * DIM) / 256, 256, 0, stream>>>(Wk, WkT);
    transpose_bf16_kernel<<<(DIM * DIM) / 256, 256, 0, stream>>>(Wv, WvT);
    transpose_bf16_kernel<<<(DIM * DIM) / 256, 256, 0, stream>>>(Wo, WoT);

    dim3 gg(DIM / 256, Mtok / 32);                // (4, 256), 4 waves per WG, 32x64 per wave
    gemm_bf16_kernel<<<gg, 128, 0, stream>>>(xb, WqT, bq, Qb,  Mtok, DIM, DIM, 0);
    gemm_bf16_kernel<<<gg, 128, 0, stream>>>(xb, WkT, bk, Kb,  Mtok, DIM, DIM, 0);
    gemm_bf16_kernel<<<gg, 128, 0, stream>>>(xb, WvT, bv, Vt,  Mtok, DIM, DIM, 1);

    attn_kernel<<<dim3(Mtok / 16), 256, 0, stream>>>(Qb, Kb, Vt, mask, Ctx);

    gemm_bf16_kernel<<<gg, 128, 0, stream>>>(Ctx, WoT, bo, d_out, Mtok, DIM, DIM, 2);
}
